// TopKGating_49160195670635
// MI455X (gfx1250) — compile-verified, matrix operand order checked by reading, not run on hardware
//
#include <hip/hip_runtime.h>
#include <hip/hip_bf16.h>
#include <math.h>

#define NTOK 16384
#define DIM  4096
#define NEXP 64
#define TOPK 8
#define TOK_PER_WG 128
#define THREADS 256

typedef __attribute__((ext_vector_type(16))) __bf16 v16bf;
typedef __attribute__((ext_vector_type(8)))  __bf16 bf16x8;
typedef __attribute__((ext_vector_type(8)))  float  v8f;

// Native converts: (__bf16)f -> v_cvt bf16 (RNE); (float)b -> exact widening.
static __device__ __forceinline__ void split_bf16(float f, __bf16& hi, __bf16& lo) {
    hi = (__bf16)f;
    lo = (__bf16)(f - (float)hi);
}

__global__ __launch_bounds__(THREADS)
void moe_gate_kernel(const float* __restrict__ x, const float* __restrict__ gw,
                     const float* __restrict__ gb, float* __restrict__ out,
                     float* __restrict__ ws_cnt, float* __restrict__ ws_disp)
{
    __shared__ float  s_logits[TOK_PER_WG * NEXP];  // 32 KB: token-major logits
    __shared__ __bf16 s_whi[2][NEXP * 32];          // 2 x 4 KB: gate_w hi slabs (double buffered)
    __shared__ __bf16 s_wlo[2][NEXP * 32];          // 2 x 4 KB: gate_w lo slabs
    __shared__ float  s_disp[NEXP];
    __shared__ float  s_cnt[NEXP];

    const int tid  = threadIdx.x;
    const int wave = tid >> 5;
    const int lane = tid & 31;
    const int half = lane >> 4;   // K-half per WMMA 16-bit layout
    const int mr   = lane & 15;   // M (A) / N (B) row within tile

    if (tid < NEXP) { s_disp[tid] = 0.0f; s_cnt[tid] = 0.0f; }

    const int rowBase = blockIdx.x * TOK_PER_WG + wave * 16;
    const float* xrow = x + (size_t)(rowBase + mr) * DIM;

    v8f acc[4];
    #pragma unroll
    for (int t = 0; t < 4; ++t) acc[t] = (v8f){0.f,0.f,0.f,0.f,0.f,0.f,0.f,0.f};

    const int we = tid >> 2;        // expert staged by this thread
    const int wk = (tid & 3) * 8;   // k offset staged by this thread
    const float* gwp = gw + (size_t)we * DIM + wk;

    // ---- prologue: stage slab for kb=0 into buffer 0
    {
        const float4 f0 = *(const float4*)(gwp);
        const float4 f1 = *(const float4*)(gwp + 4);
        float v[8] = {f0.x,f0.y,f0.z,f0.w,f1.x,f1.y,f1.z,f1.w};
        bf16x8 hi, lo;
        #pragma unroll
        for (int i = 0; i < 8; ++i) {
            __bf16 h, l;
            split_bf16(v[i], h, l);
            hi[i] = h; lo[i] = l;
        }
        *(bf16x8*)(&s_whi[0][we * 32 + wk]) = hi;
        *(bf16x8*)(&s_wlo[0][we * 32 + wk]) = lo;
    }
    __syncthreads();

    int p = 0;
    for (int kb = 0; kb < DIM; kb += 32, p ^= 1) {
        // ---- stage next slab into the other buffer while this one is consumed
        if (kb + 32 < DIM) {
            const float4 f0 = *(const float4*)(gwp + kb + 32);
            const float4 f1 = *(const float4*)(gwp + kb + 36);
            float v[8] = {f0.x,f0.y,f0.z,f0.w,f1.x,f1.y,f1.z,f1.w};
            bf16x8 hi, lo;
            #pragma unroll
            for (int i = 0; i < 8; ++i) {
                __bf16 h, l;
                split_bf16(v[i], h, l);
                hi[i] = h; lo[i] = l;
            }
            *(bf16x8*)(&s_whi[p ^ 1][we * 32 + wk]) = hi;
            *(bf16x8*)(&s_wlo[p ^ 1][we * 32 + wk]) = lo;
        }

        // ---- A fragment (16x32 bf16 layout): lanes 0-15 K 0-7/16-23, lanes 16-31 K 8-15/24-31
        const int kofs = kb + half * 8;
        const float4 a0 = *(const float4*)(xrow + kofs);
        const float4 a1 = *(const float4*)(xrow + kofs + 4);
        const float4 a2 = *(const float4*)(xrow + kofs + 16);
        const float4 a3 = *(const float4*)(xrow + kofs + 20);
        float av[16] = {a0.x,a0.y,a0.z,a0.w, a1.x,a1.y,a1.z,a1.w,
                        a2.x,a2.y,a2.z,a2.w, a3.x,a3.y,a3.z,a3.w};
        v16bf a_hi, a_lo;
        #pragma unroll
        for (int i = 0; i < 16; ++i) {
            __bf16 h, l;
            split_bf16(av[i], h, l);
            a_hi[i] = h; a_lo[i] = l;
        }

        const int lw = half * 8;
        #pragma unroll
        for (int t = 0; t < 4; ++t) {
            const int e = t * 16 + mr;  // expert = B-matrix N index
            bf16x8 h0 = *(const bf16x8*)(&s_whi[p][e * 32 + lw]);
            bf16x8 h1 = *(const bf16x8*)(&s_whi[p][e * 32 + lw + 16]);
            bf16x8 l0 = *(const bf16x8*)(&s_wlo[p][e * 32 + lw]);
            bf16x8 l1 = *(const bf16x8*)(&s_wlo[p][e * 32 + lw + 16]);
            v16bf b_hi, b_lo;
            #pragma unroll
            for (int i = 0; i < 8; ++i) {
                b_hi[i] = h0[i]; b_hi[i + 8] = h1[i];
                b_lo[i] = l0[i]; b_lo[i + 8] = l1[i];
            }
            v8f a = acc[t];
            a = __builtin_amdgcn_wmma_f32_16x16x32_bf16(false, a_hi, false, b_hi, (short)0, a, false, false);
            a = __builtin_amdgcn_wmma_f32_16x16x32_bf16(false, a_lo, false, b_hi, (short)0, a, false, false);
            a = __builtin_amdgcn_wmma_f32_16x16x32_bf16(false, a_hi, false, b_lo, (short)0, a, false, false);
            acc[t] = a;
        }
        __syncthreads();  // staging done + slab consumed before it is overwritten
    }

    // ---- bias + spill logits to LDS (C/D layout: VGPR j -> M=j+8*half, N=lane&15)
    #pragma unroll
    for (int t = 0; t < 4; ++t) {
        const int e = t * 16 + mr;
        const float bias = gb[e];
        #pragma unroll
        for (int j = 0; j < 8; ++j) {
            const int tokLoc = wave * 16 + half * 8 + j;
            s_logits[tokLoc * NEXP + e] = acc[t][j] + bias;
        }
    }
    __syncthreads();

    // ---- phase 2: one thread per token
    if (tid < TOK_PER_WG) {
        const int tokG = blockIdx.x * TOK_PER_WG + tid;
        float* row = &s_logits[tid * NEXP];

        // full softmax stats (for aux dispatch prob)
        float m = row[0];
        for (int e = 1; e < NEXP; ++e) m = fmaxf(m, row[e]);
        float ssum = 0.f;
        for (int e = 0; e < NEXP; ++e) ssum += __expf(row[e] - m);
        const float inv = 1.0f / ssum;
        for (int e = 0; e < NEXP; ++e) atomicAdd(&s_disp[e], __expf(row[e] - m) * inv);

        // top-8 selection (first-index wins on ties, like lax.top_k)
        float vsel[TOPK]; int isel[TOPK];
        #pragma unroll
        for (int k = 0; k < TOPK; ++k) {
            float best = -INFINITY; int bi = 0;
            for (int e = 0; e < NEXP; ++e) { float v = row[e]; if (v > best) { best = v; bi = e; } }
            vsel[k] = best; isel[k] = bi; row[bi] = -INFINITY;
        }
        // softmax over selected (vsel[0] is the max)
        float sw = 0.f;
        #pragma unroll
        for (int k = 0; k < TOPK; ++k) { vsel[k] = __expf(vsel[k] - vsel[0]); sw += vsel[k]; }
        const float invw = 1.0f / sw;
        float* oidx = out + (size_t)NTOK * NEXP + (size_t)tokG * TOPK;
        #pragma unroll
        for (int k = 0; k < TOPK; ++k) {
            vsel[k] *= invw;
            atomicAdd(&s_cnt[isel[k]], 1.0f);
            oidx[k] = (float)isel[k];
        }
        // dense combined-weights row: zero then scatter, then 128-bit stores
        for (int e = 0; e < NEXP; ++e) row[e] = 0.f;
        #pragma unroll
        for (int k = 0; k < TOPK; ++k) row[isel[k]] = vsel[k];
        float4* dst = (float4*)(out + (size_t)tokG * NEXP);
        const float4* srcv = (const float4*)row;
        #pragma unroll
        for (int q = 0; q < 16; ++q) dst[q] = srcv[q];
    }
    __syncthreads();
    if (tid < NEXP) {
        atomicAdd(&ws_cnt[tid],  s_cnt[tid]);
        atomicAdd(&ws_disp[tid], s_disp[tid]);
    }
}

__global__ void zero_ws_kernel(float* ws) {
    if (threadIdx.x < 2 * NEXP) ws[threadIdx.x] = 0.0f;
}

__global__ void aux_finalize_kernel(const float* __restrict__ ws_cnt,
                                    const float* __restrict__ ws_disp,
                                    float* __restrict__ out_aux) {
    __shared__ float s[NEXP];
    const int e = threadIdx.x;
    const float frac = ws_cnt[e]  * (1.0f / (float)NTOK);
    const float disp = ws_disp[e] * (1.0f / (float)NTOK);
    s[e] = frac * disp;
    __syncthreads();
    if (e == 0) {
        float t = 0.f;
        for (int i = 0; i < NEXP; ++i) t += s[i];
        *out_aux = (float)NEXP * t;
    }
}

extern "C" void kernel_launch(void* const* d_in, const int* in_sizes, int n_in,
                              void* d_out, int out_size, void* d_ws, size_t ws_size,
                              hipStream_t stream) {
    const float* x  = (const float*)d_in[0];
    const float* gw = (const float*)d_in[1];
    const float* gb = (const float*)d_in[2];
    float* out = (float*)d_out;
    float* ws_cnt  = (float*)d_ws;
    float* ws_disp = ws_cnt + NEXP;

    zero_ws_kernel<<<1, 128, 0, stream>>>((float*)d_ws);
    moe_gate_kernel<<<NTOK / TOK_PER_WG, THREADS, 0, stream>>>(x, gw, gb, out, ws_cnt, ws_disp);
    aux_finalize_kernel<<<1, NEXP, 0, stream>>>(
        ws_cnt, ws_disp, out + (size_t)NTOK * NEXP + (size_t)NTOK * TOPK);
}
